// FuzzyClass_75325136437359
// MI455X (gfx1250) — matrix operand order repaired and checked by reading, not compiled, not run on hardware
//
#include <hip/hip_runtime.h>
#include <math.h>

#define DIM_B 8
#define DIM_D 256
#define DIM_C 1024
#define DIM_R 64
#define DIM_O 256
#define NBC   (DIM_B*DIM_C)          // 8192
#define TINYF 1.17549435e-38f

typedef _Float16 v8h  __attribute__((ext_vector_type(8)));
typedef _Float16 v16h __attribute__((ext_vector_type(16)));
typedef float    v8f  __attribute__((ext_vector_type(8)));
typedef int      v4i  __attribute__((vector_size(16)));

// ---------------- workspace layout (bytes) ----------------
static constexpr size_t OFF_XH  = 0;                                      // f16 Xh[BC][D]  4 MB
static constexpr size_t OFF_WH  = OFF_XH  + (size_t)NBC*DIM_D*2;          // f16 Wh[R*O][D] 8 MB
static constexpr size_t OFF_BT  = OFF_WH  + (size_t)DIM_R*DIM_O*DIM_D*2;  // f16 bT[O][R]
static constexpr size_t OFF_A1  = OFF_BT  + (size_t)DIM_O*DIM_R*2;        // f32 center*inv_s2
static constexpr size_t OFF_A2  = OFF_A1  + (size_t)DIM_R*DIM_D*4;        // f32 inv_s2
static constexpr size_t OFF_A0  = OFF_A2  + (size_t)DIM_R*DIM_D*4;        // f32 sum c^2 inv_s2
static constexpr size_t OFF_NB  = OFF_A0  + (size_t)DIM_R*4;              // f32 norm[BC][R]
static constexpr size_t OFF_NH  = OFF_NB  + (size_t)NBC*DIM_R*4;          // f16 norm[BC][R]
static constexpr size_t OFF_IMP = OFF_NH  + (size_t)NBC*DIM_R*2;          // f32 imp[R]

// ---------------- small helpers ----------------
__device__ inline v16h ld16(const _Float16* p) {       // 16 contiguous f16
  v8h lo = *(const v8h*)p;
  v8h hi = *(const v8h*)(p + 8);
  v16h r;
#pragma unroll
  for (int i = 0; i < 8; ++i) { r[i] = lo[i]; r[i + 8] = hi[i]; }
  return r;
}
__device__ inline v16h ld8x2(const _Float16* p, int gap) { // chunks at p and p+gap
  v8h lo = *(const v8h*)p;
  v8h hi = *(const v8h*)(p + gap);
  v16h r;
#pragma unroll
  for (int i = 0; i < 8; ++i) { r[i] = lo[i]; r[i + 8] = hi[i]; }
  return r;
}

// ---- CDNA5 async global->LDS copy (ASYNCcnt path), with safe fallback ----
#if __has_builtin(__builtin_amdgcn_global_load_async_to_lds_b128)
#define HAVE_ASYNC_LDS 1
__device__ inline void cp16_async(const _Float16* g, _Float16* l) {
  __builtin_amdgcn_global_load_async_to_lds_b128((v4i*)g, (v4i*)l, 0, 0);
}
__device__ inline void cp_wait() {
  asm volatile("s_wait_asynccnt 0x0" ::: "memory");
}
#else
#define HAVE_ASYNC_LDS 0
__device__ inline void cp16_async(const _Float16* g, _Float16* l) {
  *(v8h*)l       = *(const v8h*)g;
  *(v8h*)(l + 8) = *(const v8h*)(g + 8);
}
__device__ inline void cp_wait() {}
#endif

// ---------------- K0: zero imp accumulator ----------------
__global__ void k_zero(float* imp) { imp[threadIdx.x] = 0.f; }

// ---------------- K1a: per-rule constants ----------------
__global__ __launch_bounds__(256) void k_rules(const float* __restrict__ center,
                                               const float* __restrict__ sigma,
                                               float* __restrict__ A1,
                                               float* __restrict__ A2,
                                               float* __restrict__ A0) {
  int r = blockIdx.x, d = threadIdx.x;
  float s   = sigma[r * DIM_D + d] + TINYF;
  float inv = 1.f / (s * s);
  float c   = center[r * DIM_D + d];
  A2[r * DIM_D + d] = inv;
  A1[r * DIM_D + d] = c * inv;
  __shared__ float red[256];
  red[d] = c * c * inv;
  __syncthreads();
  for (int off = 128; off > 0; off >>= 1) {
    if (d < off) red[d] += red[d + off];
    __syncthreads();
  }
  if (d == 0) A0[r] = red[0];
}

// ---------------- K1b: W -> f16 ----------------
__global__ __launch_bounds__(256) void k_cvtW(const float* __restrict__ W,
                                              _Float16* __restrict__ wh) {
  size_t i = (size_t)blockIdx.x * 256 + threadIdx.x;
  wh[i] = (_Float16)W[i];
}

// ---------------- K1c: b -> f16, transposed to (O,R) ----------------
__global__ __launch_bounds__(256) void k_cvtB(const float* __restrict__ b,
                                              _Float16* __restrict__ bt) {
  int i = blockIdx.x * 256 + threadIdx.x;   // i = o*64 + r
  int o = i >> 6, r = i & 63;
  bt[i] = (_Float16)b[r * DIM_O + o];
}

// ---------------- K1d: x (B,D,C) -> Xh (B*C, D) f16, tiled transpose -----
__global__ __launch_bounds__(256) void k_transpose(const float* __restrict__ x,
                                                   _Float16* __restrict__ xh) {
  __shared__ float t[32][33];
  int c0 = blockIdx.x * 32, d0 = blockIdx.y * 32, bb = blockIdx.z;
  int tx = threadIdx.x, ty = threadIdx.y;
#pragma unroll
  for (int i = 0; i < 4; ++i) {
    int d = d0 + ty + i * 8;
    t[ty + i * 8][tx] = x[(size_t)bb * DIM_D * DIM_C + (size_t)d * DIM_C + c0 + tx];
  }
  __syncthreads();
#pragma unroll
  for (int i = 0; i < 4; ++i) {
    int c = c0 + ty + i * 8;
    xh[((size_t)bb * DIM_C + c) * DIM_D + d0 + tx] = (_Float16)t[tx][ty + i * 8];
  }
}

// ---------------- K2: firing strengths, top-p mask, renorm, imp -----------
__global__ __launch_bounds__(64) void k_norm(const _Float16* __restrict__ xh,
                                             const float* __restrict__ A1,
                                             const float* __restrict__ A2,
                                             const float* __restrict__ A0,
                                             float tau,
                                             float* __restrict__ nb,
                                             _Float16* __restrict__ nh,
                                             float* __restrict__ imp) {
  int bc = blockIdx.x;
  int r  = threadIdx.x;
  __shared__ _Float16 xl[DIM_D];
  __shared__ float    sh[DIM_R];
  {
    const v8h* src = (const v8h*)(xh + (size_t)bc * DIM_D);
    v8h*       dst = (v8h*)xl;
    if (r < 32) dst[r] = src[r];       // 32 * 8 = 256 halves
  }
  __syncthreads();
  const float* a2 = A2 + r * DIM_D;
  const float* a1 = A1 + r * DIM_D;
  float sq = A0[r];
  for (int d = 0; d < DIM_D; ++d) {
    float xv = (float)xl[d];
    sq = fmaf(xv, fmaf(xv, a2[d], -2.f * a1[d]), sq);
  }
  float fire = __expf((-0.5f / DIM_D) * sq) + TINYF;
  sh[r] = fire; __syncthreads();
  float tot = 0.f;
  for (int j = 0; j < DIM_R; ++j) tot += sh[j];
  float nrm = fire / (tot + TINYF);
  __syncthreads(); sh[r] = nrm; __syncthreads();
  float bigger = 0.f;                               // sum of strictly larger values
  for (int j = 0; j < DIM_R; ++j) { float v = sh[j]; bigger += (v > nrm) ? v : 0.f; }
  float cs = bigger + nrm;                          // inclusive cumsum at sorted rank
  __syncthreads(); sh[r] = (cs >= tau) ? nrm : -1.f; __syncthreads();
  float vals = -1.f;                                // threshold = first value crossing tau
  for (int j = 0; j < DIM_R; ++j) vals = fmaxf(vals, sh[j]);
  float m = (nrm >= vals) ? nrm : 0.f;
  __syncthreads(); sh[r] = m; __syncthreads();
  float s2 = 0.f;
  for (int j = 0; j < DIM_R; ++j) s2 += sh[j];
  float outv = m / (s2 + TINYF);
  nb[(size_t)bc * DIM_R + r] = outv;
  nh[(size_t)bc * DIM_R + r] = (_Float16)outv;
  unsafeAtomicAdd(&imp[r], outv);
}

// ---------------- K3: fused WMMA GEMM + weighted rule reduction -----------
// WG = 8 waves; each wave owns a 16-row M tile (128 bc rows per WG), all waves
// share one 16-wide o tile. Per-rule B tile (16x256 f16, contiguous in Wh) is
// async-DMA'd into LDS double buffers and consumed by all waves.
__global__ __launch_bounds__(256) void k_pred(const _Float16* __restrict__ xh,
                                              const _Float16* __restrict__ wh,
                                              const _Float16* __restrict__ bt,
                                              const float* __restrict__ nb,
                                              const _Float16* __restrict__ nh,
                                              float* __restrict__ out) {
  const int o0     = blockIdx.x * 16;     // o tile
  const int bcBase = blockIdx.y * 128;    // bc block
  const int tid  = threadIdx.x;
  const int wave = tid >> 5;
  const int lane = tid & 31;
  const int nl   = lane & 15;
  const int half = lane >> 4;
  const int mRow = bcBase + wave * 16;    // this wave's M-tile origin

  __shared__ float    sN[128 * DIM_R];    // 32 KB norm tile
  __shared__ _Float16 sB[2][16 * DIM_D];  // 2 x 8 KB W tiles (double buffer)

  for (int i = tid; i < 128 * DIM_R; i += 256)
    sN[i] = nb[(size_t)(bcBase + (i >> 6)) * DIM_R + (i & 63)];

  // resident A tile: 16 rows x 256 K (f16), per the 16-bit A-matrix VGPR layout
  v16h A[8];
  const _Float16* xrow = xh + (size_t)(mRow + nl) * DIM_D;
#pragma unroll
  for (int kc = 0; kc < 8; ++kc)
    A[kc] = ld8x2(xrow + kc * 32 + half * 8, 16);

  // preload r = 0 B tile: contiguous 8 KB block of Wh
  {
    const _Float16* src = wh + (size_t)o0 * DIM_D;   // rows n = o0..o0+15 contiguous
#pragma unroll
    for (int k = 0; k < 2; ++k) {
      int c = tid + k * 256;                          // 512 x 16B chunks
      cp16_async(src + c * 8, &sB[0][0] + c * 8);
    }
  }
  cp_wait();
  __syncthreads();

  v8f acc = {0.f,0.f,0.f,0.f,0.f,0.f,0.f,0.f};

  for (int r = 0; r < DIM_R; ++r) {
    const int p = r & 1;
    if (r + 1 < DIM_R) {                              // prefetch next rule's B tile
      const _Float16* src = wh + ((size_t)(r + 1) * DIM_O + o0) * DIM_D;
      _Float16* dst = &sB[p ^ 1][0];
#pragma unroll
      for (int k = 0; k < 2; ++k) {
        int c = tid + k * 256;
        cp16_async(src + c * 8, dst + c * 8);
      }
    }
    // 8 chained WMMAs, B sourced from LDS
    v8f s = {0.f,0.f,0.f,0.f,0.f,0.f,0.f,0.f};
    const _Float16* bbase = &sB[p][nl * DIM_D + half * 8];
#pragma unroll
    for (int kc = 0; kc < 8; ++kc) {
      s = __builtin_amdgcn_wmma_f32_16x16x32_f16(false, A[kc], false, ld8x2(bbase + kc * 32, 16),
                                                 (short)0, s, false, false);
    }
    // weight by this rule's normalized firing strength (per output row)
    const float* sp = sN + (wave * 16 + half * 8) * DIM_R + r;
#pragma unroll
    for (int v = 0; v < 8; ++v)
      acc[v] = fmaf(sp[v * DIM_R], s[v], acc[v]);

    cp_wait();            // next buffer landed
    __syncthreads();      // everyone done reading sB[p] before it is rewritten
  }

  // bias term: acc += Norm(16xR) x bT(Rx16) -- two K=32 chained WMMAs
  const _Float16* nrow = nh + (size_t)(mRow + nl) * DIM_R;
#pragma unroll
  for (int t = 0; t < 2; ++t) {
    v16h An = ld8x2(nrow + t * 32 + half * 8, 16);
    v16h Bb = ld8x2(bt + (size_t)(o0 + nl) * DIM_R + t * 32 + half * 8, 16);
    acc = __builtin_amdgcn_wmma_f32_16x16x32_f16(false, An, false, Bb, (short)0, acc, false, false);
  }

  // store: per lane, 8 consecutive c's at row o (C/D layout puts M along VGPRs)
  const int bb    = mRow / DIM_C;
  const int cBase = mRow % DIM_C;
  float* p0 = out + ((size_t)bb * DIM_O + (o0 + nl)) * DIM_C + cBase + half * 8;
  *((float4*)p0)     = make_float4(acc[0], acc[1], acc[2], acc[3]);
  *((float4*)p0 + 1) = make_float4(acc[4], acc[5], acc[6], acc[7]);
}

// ---------------- K4: load-balance loss ----------------
__global__ __launch_bounds__(64) void k_loss(const float* __restrict__ imp,
                                             float* __restrict__ out) {
  __shared__ float sh[DIM_R];
  int t = threadIdx.x;
  sh[t] = imp[t];
  __syncthreads();
  float sum = 0.f;
  for (int j = 0; j < DIM_R; ++j) sum += sh[j];
  float mean = sum / (float)DIM_R;
  float var = 0.f;
  for (int j = 0; j < DIM_R; ++j) { float d = sh[j] - mean; var += d * d; }
  var /= (float)(DIM_R - 1);                        // ddof=1
  if (t == 0) out[(size_t)DIM_B * DIM_O * DIM_C] = 0.01f * (var / (mean * mean + 1e-20f));
}

// ---------------- launcher ----------------
extern "C" void kernel_launch(void* const* d_in, const int* in_sizes, int n_in,
                              void* d_out, int out_size, void* d_ws, size_t ws_size,
                              hipStream_t stream) {
  const float* x      = (const float*)d_in[0];
  const float* center = (const float*)d_in[1];
  const float* sigma  = (const float*)d_in[2];
  const float* W      = (const float*)d_in[3];
  const float* bvec   = (const float*)d_in[4];
  float* out = (float*)d_out;
  char*  ws  = (char*)d_ws;

  _Float16* XH = (_Float16*)(ws + OFF_XH);
  _Float16* WH = (_Float16*)(ws + OFF_WH);
  _Float16* BT = (_Float16*)(ws + OFF_BT);
  float*    A1 = (float*)(ws + OFF_A1);
  float*    A2 = (float*)(ws + OFF_A2);
  float*    A0 = (float*)(ws + OFF_A0);
  float*    NB = (float*)(ws + OFF_NB);
  _Float16* NH = (_Float16*)(ws + OFF_NH);
  float*    IMP = (float*)(ws + OFF_IMP);

  const float tau = 0.9f * powf(1.0f / (64.0f + TINYF), 1.0f / 256.0f);

  k_zero<<<1, 64, 0, stream>>>(IMP);
  k_rules<<<DIM_R, 256, 0, stream>>>(center, sigma, A1, A2, A0);
  k_cvtW<<<(DIM_R * DIM_O * DIM_D) / 256, 256, 0, stream>>>(W, WH);
  k_cvtB<<<(DIM_O * DIM_R) / 256, 256, 0, stream>>>(bvec, BT);
  k_transpose<<<dim3(DIM_C / 32, DIM_D / 32, DIM_B), dim3(32, 8), 0, stream>>>(x, XH);
  k_norm<<<NBC, 64, 0, stream>>>(XH, A1, A2, A0, tau, NB, NH, IMP);
  k_pred<<<dim3(DIM_O / 16, NBC / 128), 256, 0, stream>>>(XH, WH, BT, NB, NH, out);
  k_loss<<<1, 64, 0, stream>>>(IMP, out);
}
